// GATEmbedding_20684562498294
// MI455X (gfx1250) — matrix operand order repaired and checked by reading, not compiled, not run on hardware
//
#include <hip/hip_runtime.h>
#include <hip/hip_bf16.h>
#include <math.h>

// Problem constants (match reference)
#define NN   50000
#define EEc  400000
#define EAc  (EEc + NN)   // extended edges incl. self loops
#define C100 100
#define EDIM 200          // MSG_DIM + TIME_DIM
#define NEG_SLOPE 0.2f

// padded sizes for WMMA staging
#define KP100 128         // K=100 padded to mult of 32
#define KP200 224         // K=200 padded to mult of 32
#define EApad 450064      // EAc padded to mult of 16 (NN already mult of 16)

// order-preserving encoding of -INFINITY (bits 0xFF800000, negative -> ~bits)
#define FENC_NEG_INF 0x007FFFFFu

typedef __attribute__((ext_vector_type(16))) _Float16 v16h;
typedef __attribute__((ext_vector_type(8)))  _Float16 v8h;
typedef __attribute__((ext_vector_type(8)))  float    v8f;

// ---------- helpers ----------
__device__ __forceinline__ void edge_nodes(const int* src, const int* dst, int e,
                                           int& s, int& d) {
  if (e < EEc) { s = src[e]; d = dst[e]; }
  else         { s = d = e - EEc; }
}
// order-preserving float<->uint encoding for atomic max on floats
__device__ __forceinline__ unsigned fenc(float f) {
  unsigned u = __float_as_uint(f);
  return (u & 0x80000000u) ? ~u : (u | 0x80000000u);
}
__device__ __forceinline__ float fdec(unsigned u) {
  unsigned b = (u & 0x80000000u) ? (u & 0x7FFFFFFFu) : ~u;
  return __uint_as_float(b);
}

// ---------- utility kernels ----------
__global__ void fill_f32(float* p, float v, size_t n) {
  size_t i = (size_t)blockIdx.x * blockDim.x + threadIdx.x;
  if (i < n) p[i] = v;
}
__global__ void fill_u32(unsigned* p, unsigned v, size_t n) {
  size_t i = (size_t)blockIdx.x * blockDim.x + threadIdx.x;
  if (i < n) p[i] = v;
}
// fp32 [rows,K] -> f16 [rowsPad,Kpad], zero padded (rows beyond 'rows' and k>=K)
__global__ void cvt_pad_f16(const float* __restrict__ in, _Float16* __restrict__ out,
                            int rows, int K, int rowsPad, int Kpad) {
  size_t tid = (size_t)blockIdx.x * blockDim.x + threadIdx.x;
  size_t total = (size_t)rowsPad * Kpad;
  if (tid >= total) return;
  int r = (int)(tid / Kpad);
  int k = (int)(tid % Kpad);
  out[tid] = (r < rows && k < K) ? (_Float16)in[(size_t)r * K + k] : (_Float16)0.0f;
}

// ---------- pack fp32 weight W[K,Nc] into WMMA B-fragment tiles ----------
// Layout: [ntile][ktile][lane][16 halves]; lane -> col = ntile*16 + (lane&15),
// K rows kb..kb+15 with kb = ktile*32 + (lane>=16 ? 16 : 0). Zero padded.
// Buffer is over-allocated by one column tile (dummy) so the GEMM's second-tile
// load needs no guard.
__global__ void pack_b_f16(const float* __restrict__ W, _Float16* __restrict__ Bp,
                           int K, int Nc, int Kpad) {
  int nt = (Nc + 15) >> 4;
  int kt = Kpad >> 5;
  size_t tid = (size_t)blockIdx.x * blockDim.x + threadIdx.x;
  size_t total = (size_t)nt * kt * 32;
  if (tid >= total) return;
  int lane = (int)(tid & 31);
  size_t tile = tid >> 5;
  int ktile = (int)(tile % kt);
  int ntile = (int)(tile / kt);
  int col = ntile * 16 + (lane & 15);
  int kb = ktile * 32 + ((lane >= 16) ? 16 : 0);
  _Float16* o = Bp + tile * 512 + (size_t)lane * 16;
#pragma unroll
  for (int i = 0; i < 16; ++i) {
    int k = kb + i;
    o[i] = (k < K && col < Nc) ? (_Float16)W[(size_t)k * Nc + col] : (_Float16)0.0f;
  }
}

// ---------- edge_attr construction ----------
// ea rows [0,E): cos(rel_t*Wt+bt) ++ msg ; rows [E,E+N): atomic sum for self-loop mean
__global__ void build_edge_attr(const float* last_update, const float* t,
                                const float* msg, const int* src, const int* dst,
                                const float* Wt, const float* bt,
                                float* ea, float* cnt) {
  size_t tid = (size_t)blockIdx.x * blockDim.x + threadIdx.x;
  if (tid >= (size_t)EEc * EDIM) return;
  int e = (int)(tid / EDIM);
  int d = (int)(tid % EDIM);
  int s  = src[e];
  int dd = dst[e];
  float v;
  if (d < C100) {
    float rel = last_update[s] - t[e];
    v = cosf(rel * Wt[d] + bt[d]);
  } else {
    v = msg[(size_t)e * C100 + (d - C100)];
  }
  ea[(size_t)e * EDIM + d] = v;
  atomicAdd(&ea[((size_t)EEc + dd) * EDIM + d], v);
  if (d == 0) atomicAdd(&cnt[dd], 1.0f);
}

__global__ void finalize_loop_rows(float* ea, const float* cnt) {
  size_t tid = (size_t)blockIdx.x * blockDim.x + threadIdx.x;
  if (tid >= (size_t)NN * EDIM) return;
  int n = (int)(tid / EDIM);
  ea[(size_t)EEc * EDIM + tid] /= fmaxf(cnt[n], 1.0f);
}

// ---------- WMMA GEMM ----------
// A: f16 [Mpad, Kpad] zero-padded. Bp: packed fragment tiles (+1 dummy tile).
// Each wave computes a 16x32 strip of C (one A fragment reused for 2 B tiles).
// Inner loop is branchless: 2 b128 A loads + 4 b128 B loads + 2 WMMAs; the
// second WMMA's result is discarded at store time if its column tile is
// out of range (EXEC stays all-ones as WMMA requires).
__global__ void wmma_gemm_f16(const _Float16* __restrict__ A,
                              const _Float16* __restrict__ Bp,
                              const float* __restrict__ bias,
                              float* __restrict__ C,
                              int M, int Nc, int Kpad) {
  int wid  = (int)(((size_t)blockIdx.x * blockDim.x + threadIdx.x) >> 5);
  int lane = threadIdx.x & 31;
  int mt = (M + 15) >> 4;
  int nt = (Nc + 15) >> 4;
  int np = (nt + 1) >> 1;                 // column-tile pairs
  if (wid >= mt * np) return;
  int mtile = wid / np;
  int npair = wid % np;
  int m0 = mtile * 16;
  int n0 = npair * 32;
  int kt = Kpad >> 5;
  bool hiHalf = lane >= 16;
  int row = m0 + (lane & 15);

  const _Float16* ap  = A + (size_t)row * Kpad + (hiHalf ? 16 : 0);
  const _Float16* bp0 = Bp + (size_t)(npair * 2) * kt * 512 + (size_t)lane * 16;
  const _Float16* bp1 = bp0 + (size_t)kt * 512;   // may point at dummy tile

  v8f acc0 = {};
  v8f acc1 = {};
  for (int k = 0; k < kt; ++k) {
    __builtin_prefetch(ap + 32, 0, 3);       // next A K-slab, keep near (WGP)
    v8h lo = *(const v8h*)(ap);
    v8h hh = *(const v8h*)(ap + 8);
    ap += 32;
    v16h b0 = *(const v16h*)(bp0);
    v16h b1 = *(const v16h*)(bp1);
    bp0 += 512;
    bp1 += 512;
    v16h a;
#pragma unroll
    for (int i = 0; i < 8; ++i) { a[i] = lo[i]; a[i + 8] = hh[i]; }
    acc0 = __builtin_amdgcn_wmma_f32_16x16x32_f16(
        false, a, false, b0, (short)0, acc0, false, false);
    acc1 = __builtin_amdgcn_wmma_f32_16x16x32_f16(
        false, a, false, b1, (short)0, acc1, false, false);
  }
  // C/D f32 layout: VGPR r, lane l -> M = m0 + r + (l>=16 ? 8 : 0), N = n0 + (l&15)
  int rbase = m0 + (hiHalf ? 8 : 0);
  int col0 = n0 + (lane & 15);
  int col1 = col0 + 16;
  if (col0 < Nc) {
    float bv = bias ? bias[col0] : 0.0f;
#pragma unroll
    for (int r = 0; r < 8; ++r) {
      int m = rbase + r;
      if (m < M) C[(size_t)m * Nc + col0] = acc0[r] + bv;
    }
  }
  if (col1 < Nc) {
    float bv = bias ? bias[col1] : 0.0f;
#pragma unroll
    for (int r = 0; r < 8; ++r) {
      int m = rbase + r;
      if (m < M) C[(size_t)m * Nc + col1] = acc1[r] + bv;
    }
  }
}

// ---------- GATv2 attention (3-pass segment softmax) ----------
// Pass A: per-edge logits, atomic segment max. One wave per extended edge.
__global__ void attn_logit(const float* __restrict__ xl, const float* __restrict__ xr,
                           const float* __restrict__ ee, const float* __restrict__ att,
                           float* logit, unsigned* amax,
                           const int* src, const int* dst, int H) {
  int wid  = (int)(((size_t)blockIdx.x * blockDim.x + threadIdx.x) >> 5);
  int lane = threadIdx.x & 31;
  if (wid >= EAc) return;
  int s, d; edge_nodes(src, dst, wid, s, d);
  int HC = H * C100;
  float p0 = 0.0f, p1 = 0.0f;
  for (int c = lane; c < HC; c += 32) {
    float v = xl[(size_t)s * HC + c] + xr[(size_t)d * HC + c] +
              ee[(size_t)wid * HC + c];
    v = (v > 0.0f) ? v : NEG_SLOPE * v;
    float pv = v * att[c];          // att is (H,100) row-major -> index == c
    if (c < C100) p0 += pv; else p1 += pv;
  }
#pragma unroll
  for (int o = 16; o; o >>= 1) {
    p0 += __shfl_xor(p0, o, 32);
    p1 += __shfl_xor(p1, o, 32);
  }
  if (lane == 0) {
    logit[(size_t)wid * H] = p0;
    atomicMax(&amax[(size_t)d * H], fenc(p0));
    if (H == 2) {
      logit[(size_t)wid * H + 1] = p1;
      atomicMax(&amax[(size_t)d * H + 1], fenc(p1));
    }
  }
}

// Pass B: exp(logit - max), accumulate denominators. Overwrites logit with ex.
__global__ void attn_exp(float* logit, const unsigned* amax, float* den,
                         const int* src, const int* dst, int H) {
  size_t tid = (size_t)blockIdx.x * blockDim.x + threadIdx.x;
  if (tid >= (size_t)EAc * H) return;
  int e = (int)(tid / H);
  int h = (int)(tid % H);
  int s, d; edge_nodes(src, dst, e, s, d);
  float ex = expf(logit[tid] - fdec(amax[(size_t)d * H + h]));
  logit[tid] = ex;
  atomicAdd(&den[(size_t)d * H + h], ex);
}

// Pass C: acc[dst] += xl[src] * alpha
__global__ void attn_agg(const float* __restrict__ xl, const float* __restrict__ ex,
                         const float* __restrict__ den, float* acc,
                         const int* src, const int* dst, int H) {
  size_t tid = (size_t)blockIdx.x * blockDim.x + threadIdx.x;
  int HC = H * C100;
  if (tid >= (size_t)EAc * HC) return;
  int e = (int)(tid / HC);
  int i = (int)(tid % HC);
  int h = i / C100;
  int s, d; edge_nodes(src, dst, e, s, d);
  float alpha = ex[(size_t)e * H + h] / den[(size_t)d * H + h];
  atomicAdd(&acc[(size_t)d * HC + i], xl[(size_t)s * HC + i] * alpha);
}

__global__ void finalize_relu_bias(float* acc, const float* bias, int HC, size_t n) {
  size_t tid = (size_t)blockIdx.x * blockDim.x + threadIdx.x;
  if (tid >= n) return;
  float v = acc[tid] + bias[tid % HC];
  acc[tid] = v > 0.0f ? v : 0.0f;
}
__global__ void finalize_out(const float* acc, const float* bias, float* out, size_t n) {
  size_t tid = (size_t)blockIdx.x * blockDim.x + threadIdx.x;
  if (tid >= n) return;
  out[tid] = acc[tid] + bias[tid % C100];
}

// ---------- host orchestration ----------
extern "C" void kernel_launch(void* const* d_in, const int* in_sizes, int n_in,
                              void* d_out, int out_size, void* d_ws, size_t ws_size,
                              hipStream_t stream) {
  (void)in_sizes; (void)n_in; (void)out_size; (void)ws_size;
  const float* x      = (const float*)d_in[0];
  const float* lastup = (const float*)d_in[1];
  const float* t      = (const float*)d_in[2];
  const float* msg    = (const float*)d_in[3];
  const int*   ei     = (const int*)d_in[4];       // [2,E]
  const float* Wt  = (const float*)d_in[5];
  const float* bt  = (const float*)d_in[6];
  const float* Wl1 = (const float*)d_in[7];
  const float* bl1 = (const float*)d_in[8];
  const float* Wr1 = (const float*)d_in[9];
  const float* br1 = (const float*)d_in[10];
  const float* We1 = (const float*)d_in[11];
  const float* att1= (const float*)d_in[12];
  const float* b1  = (const float*)d_in[13];
  const float* Wl2 = (const float*)d_in[14];
  const float* bl2 = (const float*)d_in[15];
  const float* Wr2 = (const float*)d_in[16];
  const float* br2 = (const float*)d_in[17];
  const float* We2 = (const float*)d_in[18];
  const float* att2= (const float*)d_in[19];
  const float* b2  = (const float*)d_in[20];
  const int* srcp = ei;
  const int* dstp = ei + EEc;
  float* out = (float*)d_out;

  // workspace layout
  char* ws = (char*)d_ws;
  size_t off = 0;
  auto alloc = [&](size_t bytes) -> void* {
    void* p = ws + off;
    off = (off + bytes + 255) & ~(size_t)255;
    return p;
  };
  float*    ea    = (float*)   alloc((size_t)EAc * EDIM * 4);
  _Float16* eah   = (_Float16*)alloc((size_t)EApad * KP200 * 2);
  _Float16* xh    = (_Float16*)alloc((size_t)NN * KP200 * 2);
  float*    cnt   = (float*)   alloc((size_t)NN * 4);
  float*    xl    = (float*)   alloc((size_t)NN * EDIM * 4);
  float*    xr    = (float*)   alloc((size_t)NN * EDIM * 4);
  float*    ee    = (float*)   alloc((size_t)EAc * EDIM * 4);
  float*    logit = (float*)   alloc((size_t)EAc * 2 * 4);
  unsigned* amax  = (unsigned*)alloc((size_t)NN * 2 * 4);
  float*    den   = (float*)   alloc((size_t)NN * 2 * 4);
  float*    acc   = (float*)   alloc((size_t)NN * EDIM * 4);
  // packed B fragment buffers: (nt+1) * kt * 32 lanes * 16 halves
  // (+1 column tile of dummy space so second-tile loads need no guard)
  _Float16* wl1p = (_Float16*)alloc((size_t)14 * 4 * 512 * 2);  // K=100,Nc=200
  _Float16* wr1p = (_Float16*)alloc((size_t)14 * 4 * 512 * 2);
  _Float16* we1p = (_Float16*)alloc((size_t)14 * 7 * 512 * 2);  // K=200,Nc=200
  _Float16* wl2p = (_Float16*)alloc((size_t)8  * 7 * 512 * 2);  // K=200,Nc=100
  _Float16* wr2p = (_Float16*)alloc((size_t)8  * 7 * 512 * 2);
  _Float16* we2p = (_Float16*)alloc((size_t)8  * 7 * 512 * 2);

  const int B = 256;
  auto blocks = [&](size_t n) { return (unsigned)((n + B - 1) / B); };
  auto gemm = [&](const _Float16* A, const _Float16* Bpk, const float* bias,
                  float* Cc, int M, int Nc, int Kpad) {
    int mt = (M + 15) / 16;
    int np = ((Nc + 15) / 16 + 1) / 2;
    size_t waves = (size_t)mt * np;
    wmma_gemm_f16<<<blocks(waves * 32), B, 0, stream>>>(A, Bpk, bias, Cc, M, Nc, Kpad);
  };
  auto packb = [&](const float* W, _Float16* Bpk, int K, int Nc, int Kpad) {
    size_t n = (size_t)((Nc + 15) / 16) * (Kpad / 32) * 32;
    pack_b_f16<<<blocks(n), B, 0, stream>>>(W, Bpk, K, Nc, Kpad);
  };

  // ---- edge_attr + self-loop mean rows ----
  fill_f32<<<blocks((size_t)NN * EDIM), B, 0, stream>>>(ea + (size_t)EEc * EDIM, 0.0f,
                                                        (size_t)NN * EDIM);
  fill_f32<<<blocks(NN), B, 0, stream>>>(cnt, 0.0f, (size_t)NN);
  build_edge_attr<<<blocks((size_t)EEc * EDIM), B, 0, stream>>>(
      lastup, t, msg, srcp, dstp, Wt, bt, ea, cnt);
  finalize_loop_rows<<<blocks((size_t)NN * EDIM), B, 0, stream>>>(ea, cnt);

  // ---- staging conversions (padded) + weight packing ----
  cvt_pad_f16<<<blocks((size_t)EApad * KP200), B, 0, stream>>>(ea, eah, EAc, EDIM,
                                                               EApad, KP200);
  cvt_pad_f16<<<blocks((size_t)NN * KP100), B, 0, stream>>>(x, xh, NN, C100, NN, KP100);
  packb(Wl1, wl1p, C100, 200, KP100);
  packb(Wr1, wr1p, C100, 200, KP100);
  packb(We1, we1p, EDIM, 200, KP200);
  packb(Wl2, wl2p, EDIM, C100, KP200);
  packb(Wr2, wr2p, EDIM, C100, KP200);
  packb(We2, we2p, EDIM, C100, KP200);

  // ================= layer 1 (heads=2, concat) =================
  gemm(xh,  wl1p, bl1, xl, NN,  200, KP100);
  gemm(xh,  wr1p, br1, xr, NN,  200, KP100);
  gemm(eah, we1p, 0,   ee, EAc, 200, KP200);

  fill_u32<<<blocks((size_t)NN * 2), B, 0, stream>>>(amax, FENC_NEG_INF, (size_t)NN * 2);
  fill_f32<<<blocks((size_t)NN * 2), B, 0, stream>>>(den, 0.0f, (size_t)NN * 2);
  fill_f32<<<blocks((size_t)NN * 200), B, 0, stream>>>(acc, 0.0f, (size_t)NN * 200);

  attn_logit<<<blocks((size_t)EAc * 32), B, 0, stream>>>(xl, xr, ee, att1, logit, amax,
                                                         srcp, dstp, 2);
  attn_exp<<<blocks((size_t)EAc * 2), B, 0, stream>>>(logit, amax, den, srcp, dstp, 2);
  attn_agg<<<blocks((size_t)EAc * 200), B, 0, stream>>>(xl, logit, den, acc, srcp, dstp, 2);
  finalize_relu_bias<<<blocks((size_t)NN * 200), B, 0, stream>>>(acc, b1, 200,
                                                                 (size_t)NN * 200);
  cvt_pad_f16<<<blocks((size_t)NN * KP200), B, 0, stream>>>(acc, xh, NN, EDIM, NN, KP200);

  // ================= layer 2 (heads=1, mean) =================
  gemm(xh,  wl2p, bl2, xl, NN,  C100, KP200);
  gemm(xh,  wr2p, br2, xr, NN,  C100, KP200);
  gemm(eah, we2p, 0,   ee, EAc, C100, KP200);

  fill_u32<<<blocks(NN), B, 0, stream>>>(amax, FENC_NEG_INF, (size_t)NN);
  fill_f32<<<blocks(NN), B, 0, stream>>>(den, 0.0f, (size_t)NN);
  fill_f32<<<blocks((size_t)NN * C100), B, 0, stream>>>(acc, 0.0f, (size_t)NN * C100);

  attn_logit<<<blocks((size_t)EAc * 32), B, 0, stream>>>(xl, xr, ee, att2, logit, amax,
                                                         srcp, dstp, 1);
  attn_exp<<<blocks((size_t)EAc), B, 0, stream>>>(logit, amax, den, srcp, dstp, 1);
  attn_agg<<<blocks((size_t)EAc * C100), B, 0, stream>>>(xl, logit, den, acc, srcp, dstp, 1);
  finalize_out<<<blocks((size_t)NN * C100), B, 0, stream>>>(acc, b2, out,
                                                            (size_t)NN * C100);
}
